// Attention_43198781063919
// MI455X (gfx1250) — matrix operand order repaired and checked by reading, not compiled, not run on hardware
//
#include <hip/hip_runtime.h>
#include <hip/hip_bf16.h>

#define B_      32
#define L_      512
#define H2_     1024
#define OUTD_   6
#define TOPK_   10
#define ITILE_  64          // rows of scores per block
#define NBLK2_  (B_ * OUTD_ * (L_ / ITILE_))   // 1536 fused blocks
#define SMEM_BYTES (ITILE_ * H2_ * 2 + 2560 * 4 + 256 * 4 + 256 * 4) // 143360

typedef __bf16 bf16_t;
typedef __attribute__((ext_vector_type(16))) __bf16 v16bf;
typedef __attribute__((ext_vector_type(8)))  float  v8f;

__device__ __forceinline__ unsigned short f32_to_bf16_rne(float f) {
    unsigned int u = __float_as_uint(f);
    u += 0x7FFFu + ((u >> 16) & 1u);
    return (unsigned short)(u >> 16);
}

// ---- WMMA fragment loaders (bf16, 16x16x32 shapes) -------------------------
// A (16x32, row-major, ld in halves): lane l<16 -> row l, K=0..7 & 16..23;
// lane l>=16 -> row l-16, K=8..15 & 24..31.
__device__ __forceinline__ v16bf load_frag_a(const unsigned short* p, int row0,
                                             int k0, int ld, int lane) {
    int r  = lane & 15;
    int hi = lane >> 4;
    const unsigned short* q = p + (size_t)(row0 + r) * ld + k0 + hi * 8;
    union { uint4 q2[2]; v16bf v; } u;
    u.q2[0] = *reinterpret_cast<const uint4*>(q);
    u.q2[1] = *reinterpret_cast<const uint4*>(q + 16);
    return u.v;
}

// B (32x16): lane l<16 -> col l, K=0..7 & 16..23; lane l>=16 -> col l-16, K=8..15 & 24..31.
// Columns of B are rows of the transposed-storage operand (contig in K).
__device__ __forceinline__ v16bf load_frag_b(const unsigned short* p, int col0,
                                             int k0, int ld, int lane) {
    int c  = lane & 15;
    int hi = lane >> 4;
    const unsigned short* q = p + (size_t)(col0 + c) * ld + k0 + hi * 8;
    union { uint4 q2[2]; v16bf v; } u;
    u.q2[0] = *reinterpret_cast<const uint4*>(q);
    u.q2[1] = *reinterpret_cast<const uint4*>(q + 16);
    return u.v;
}

__device__ __forceinline__ v8f wmma_bf16(v16bf a, v16bf b, v8f c) {
    return __builtin_amdgcn_wmma_f32_16x16x32_bf16(
        false, a, false, b, (short)0, c, false, false);
}

// C/D 16x16 f32 layout: lane l<16 -> N=l, M=vgpr; lane l>=16 -> N=l-16, M=vgpr+8.
__device__ __forceinline__ void store_acc_lds(unsigned short* Ut, int m0, int n0,
                                              v8f acc, int lane) {
    int c   = lane & 15;
    int hi  = lane >> 4;
    int col = n0 + c;
    int mb  = m0 + hi * 8;
#pragma unroll
    for (int v = 0; v < 8; ++v)
        Ut[(size_t)(mb + v) * H2_ + col] = f32_to_bf16_rne(acc[v]);
}

// ---- register top-k (sorted descending, fully unrolled, no dyn indexing) ---
__device__ __forceinline__ void topk_insert(float (&tk)[TOPK_], float v) {
    if (v <= tk[TOPK_ - 1]) return;
    tk[TOPK_ - 1] = v;
#pragma unroll
    for (int s = TOPK_ - 1; s >= 1; --s) {
        float a = tk[s - 1], b = tk[s];
        tk[s - 1] = fmaxf(a, b);
        tk[s]     = fminf(a, b);
    }
}

// ============================================================================
// Kernel 1: f32 -> bf16 for sent1/sent2; W transposed to WT[o][q][p] (bf16)
// ============================================================================
__global__ __launch_bounds__(256) void convert_kernel(
    const float* __restrict__ s1, const float* __restrict__ s2,
    const float* __restrict__ W, unsigned short* __restrict__ s1b,
    unsigned short* __restrict__ s2b, unsigned short* __restrict__ WT) {
    const int stride = gridDim.x * blockDim.x;
    const int N  = B_ * L_ * H2_;           // 16,777,216
    const int NW = OUTD_ * H2_ * H2_;       //  6,291,456
    for (int i = blockIdx.x * blockDim.x + threadIdx.x; i < N; i += stride) {
        s1b[i] = f32_to_bf16_rne(s1[i]);
        s2b[i] = f32_to_bf16_rne(s2[i]);
    }
    for (int i = blockIdx.x * blockDim.x + threadIdx.x; i < NW; i += stride) {
        int o = i >> 20;            // H2*H2 == 1<<20
        int r = i & 0xFFFFF;
        int p = r >> 10;
        int q = r & 1023;
        WT[((size_t)o << 20) + ((size_t)q << 10) + p] = f32_to_bf16_rne(W[i]);
    }
}

// ============================================================================
// Kernel 2: fused  U = S1_b * W_o  (LDS-resident, bf16)   then
//           scores = U * S2_b^T   folded straight into block top-10
// one block per (b, o, itile of 64 rows)
// ============================================================================
__global__ __launch_bounds__(256) void fused_gemm_topk(
    const unsigned short* __restrict__ s1b, const unsigned short* __restrict__ s2b,
    const unsigned short* __restrict__ WT, float* __restrict__ partial) {
    extern __shared__ char smem[];
    unsigned short* Ut = (unsigned short*)smem;                     // 64x1024 bf16
    float* cand = (float*)(smem + ITILE_ * H2_ * 2);                // 2560 f32
    float* redv = (float*)((char*)cand + 2560 * 4);                 // 256 f32
    int*   redi = (int*)((char*)redv + 256 * 4);                    // 256 i32

    const int tid  = threadIdx.x;
    const int lane = tid & 31;
    const int wave = tid >> 5;
    const int blk  = blockIdx.x;
    const int b     = blk / (OUTD_ * (L_ / ITILE_));
    const int rem   = blk % (OUTD_ * (L_ / ITILE_));
    const int o     = rem >> 3;
    const int itile = rem & 7;
    const int i0    = itile * ITILE_;

    const unsigned short* A1 = s1b + ((size_t)b * L_ + i0) * H2_;   // 64 x 1024
    const unsigned short* B1 = WT + ((size_t)o << 20);              // q-major W_o

    // ---- stage 1: Ut[m][q] = sum_p A1[m][p] * W[o][p][q] ----
    // wave tile: 64 (all rows) x 64 (q); 16 n-tiles, 2 per wave
    for (int t = wave; t < H2_ / 64; t += 8) {
        const int n0 = t * 64;
        v8f acc[4][4];
#pragma unroll
        for (int mi = 0; mi < 4; ++mi)
#pragma unroll
            for (int ni = 0; ni < 4; ++ni) acc[mi][ni] = (v8f){};
        for (int k0 = 0; k0 < H2_; k0 += 32) {
            v16bf fa[4], fb[4];
#pragma unroll
            for (int mi = 0; mi < 4; ++mi)
                fa[mi] = load_frag_a(A1, mi * 16, k0, H2_, lane);
#pragma unroll
            for (int ni = 0; ni < 4; ++ni)
                fb[ni] = load_frag_b(B1, n0 + ni * 16, k0, H2_, lane);
#pragma unroll
            for (int mi = 0; mi < 4; ++mi)
#pragma unroll
                for (int ni = 0; ni < 4; ++ni)
                    acc[mi][ni] = wmma_bf16(fa[mi], fb[ni], acc[mi][ni]);
        }
#pragma unroll
        for (int mi = 0; mi < 4; ++mi)
#pragma unroll
            for (int ni = 0; ni < 4; ++ni)
                store_acc_lds(Ut, mi * 16, n0 + ni * 16, acc[mi][ni], lane);
    }
    __syncthreads();

    // ---- stage 2: scores[m][j] = sum_q Ut[m][q] * s2[b][j][q], fold top-k ----
    // wave tile: 64 (all rows, LDS A) x 32 (j, global B); 16 j-tiles, 2 per wave
    float tk[TOPK_];
#pragma unroll
    for (int t = 0; t < TOPK_; ++t) tk[t] = -__builtin_huge_valf();

    const unsigned short* B2 = s2b + (size_t)b * L_ * H2_;          // 512 rows
    for (int t = wave; t < L_ / 32; t += 8) {
        const int j0 = t * 32;
        v8f acc[4][2];
#pragma unroll
        for (int mi = 0; mi < 4; ++mi)
#pragma unroll
            for (int ji = 0; ji < 2; ++ji) acc[mi][ji] = (v8f){};
        for (int k0 = 0; k0 < H2_; k0 += 32) {
            v16bf fa[4], fb[2];
#pragma unroll
            for (int mi = 0; mi < 4; ++mi)
                fa[mi] = load_frag_a(Ut, mi * 16, k0, H2_, lane);   // LDS
#pragma unroll
            for (int ji = 0; ji < 2; ++ji)
                fb[ji] = load_frag_b(B2, j0 + ji * 16, k0, H2_, lane);
#pragma unroll
            for (int mi = 0; mi < 4; ++mi)
#pragma unroll
                for (int ji = 0; ji < 2; ++ji)
                    acc[mi][ji] = wmma_bf16(fa[mi], fb[ji], acc[mi][ji]);
        }
#pragma unroll
        for (int mi = 0; mi < 4; ++mi)
#pragma unroll
            for (int ji = 0; ji < 2; ++ji)
#pragma unroll
                for (int v = 0; v < 8; ++v) topk_insert(tk, acc[mi][ji][v]);
    }

    // ---- block reduction: repeated argmax over 256*10 candidates ----
#pragma unroll
    for (int t = 0; t < TOPK_; ++t) cand[tid * TOPK_ + t] = tk[t];
    __syncthreads();

    for (int r = 0; r < TOPK_; ++r) {
        float m = -__builtin_huge_valf();
        int   mi = tid * TOPK_;
        for (int t = 0; t < TOPK_; ++t) {
            float c = cand[tid * TOPK_ + t];
            if (c > m) { m = c; mi = tid * TOPK_ + t; }
        }
        redv[tid] = m;
        redi[tid] = mi;
        __syncthreads();
        for (int s = 128; s > 0; s >>= 1) {
            if (tid < s && redv[tid + s] > redv[tid]) {
                redv[tid] = redv[tid + s];
                redi[tid] = redi[tid + s];
            }
            __syncthreads();
        }
        if (tid == 0) {
            partial[(size_t)blk * TOPK_ + r] = redv[0];
            cand[redi[0]] = -__builtin_huge_valf();
        }
        __syncthreads();
    }
}

// ============================================================================
// Kernel 3: merge the 8 per-itile partial lists of each (b,o) -> d_out
// ============================================================================
__global__ __launch_bounds__(256) void final_topk(const float* __restrict__ partial,
                                                  float* __restrict__ out) {
    int g = blockIdx.x * blockDim.x + threadIdx.x;
    if (g >= B_ * OUTD_) return;
    float tk[TOPK_];
#pragma unroll
    for (int t = 0; t < TOPK_; ++t) tk[t] = -__builtin_huge_valf();
    const float* p = partial + (size_t)g * (L_ / ITILE_) * TOPK_;
    for (int i = 0; i < (L_ / ITILE_) * TOPK_; ++i) topk_insert(tk, p[i]);
#pragma unroll
    for (int t = 0; t < TOPK_; ++t) out[(size_t)g * TOPK_ + t] = tk[t];
}

// ============================================================================
extern "C" void kernel_launch(void* const* d_in, const int* in_sizes, int n_in,
                              void* d_out, int out_size, void* d_ws, size_t ws_size,
                              hipStream_t stream) {
    const float* s1 = (const float*)d_in[0];
    const float* s2 = (const float*)d_in[1];
    const float* W  = (const float*)d_in[2];
    float* out = (float*)d_out;

    char* ws = (char*)d_ws;
    size_t off = 0;
    unsigned short* s1b = (unsigned short*)(ws + off); off += (size_t)B_ * L_ * H2_ * 2;
    unsigned short* s2b = (unsigned short*)(ws + off); off += (size_t)B_ * L_ * H2_ * 2;
    unsigned short* WT  = (unsigned short*)(ws + off); off += (size_t)OUTD_ * H2_ * H2_ * 2;
    float* partial      = (float*)(ws + off);          off += (size_t)NBLK2_ * TOPK_ * 4;

    (void)hipFuncSetAttribute((const void*)fused_gemm_topk,
                              hipFuncAttributeMaxDynamicSharedMemorySize,
                              SMEM_BYTES);

    convert_kernel<<<4096, 256, 0, stream>>>(s1, s2, W, s1b, s2b, WT);
    fused_gemm_topk<<<NBLK2_, 256, SMEM_BYTES, stream>>>(s1b, s2b, WT, partial);
    final_topk<<<1, 256, 0, stream>>>(partial, out);
}
